// EfficientEntangledAttention_26542897889391
// MI455X (gfx1250) — matrix operand order repaired and checked by reading, not compiled, not run on hardware
//
#include <hip/hip_runtime.h>
#include <hip/hip_bf16.h>

// ---------------- problem constants ----------------
constexpr int B_  = 4;
constexpr int T_  = 4096;
constexpr int D_  = 1024;   // DIM
constexpr int H_  = 16;     // heads
constexpr int HD_ = 64;     // head dim
constexpr int R_  = 16;     // rank
constexpr int BT_ = B_ * T_;
constexpr int HR_ = H_ * R_;  // 256 = folded K of final GEMM

typedef __bf16 bf16;
typedef __attribute__((ext_vector_type(16))) __bf16 bf16x16;
typedef __attribute__((ext_vector_type(8)))  float  floatx8;
typedef __attribute__((ext_vector_type(4)))  unsigned int u32x4;
typedef __attribute__((ext_vector_type(4)))  float  f32x4;

__device__ __forceinline__ bf16  f2bf(float v) { return (bf16)v; }
__device__ __forceinline__ float bf2f(bf16 v)  { return (float)v; }

__device__ __forceinline__ floatx8 zero8() {
  floatx8 v;
#pragma unroll
  for (int i = 0; i < 8; ++i) v[i] = 0.0f;
  return v;
}

// WMMA D = A(16x32 bf16) * B(32x16 bf16) + C(f32)
__device__ __forceinline__ floatx8 wmma_bf16(bf16x16 a, bf16x16 b, floatx8 c) {
  return __builtin_amdgcn_wmma_f32_16x16x32_bf16(
      false, a, false, b, (short)0, c, false, false);
}

// Build a fragment from two contiguous 16-byte chunks (must be 16B aligned).
__device__ __forceinline__ bf16x16 load_frag_2x16(const bf16* p0, const bf16* p1) {
  union { bf16x16 v; u32x4 q[2]; } u;
  u.q[0] = *reinterpret_cast<const u32x4*>(p0);
  u.q[1] = *reinterpret_cast<const u32x4*>(p1);
  return u.v;
}

// A fragment: 16x32 (MxK) from row-major LDS tile, leading dim lda (elements).
// ISA: lanes 0-15 -> M=lane, K{0..7,16..23}; lanes 16-31 -> M=lane-16, K{8..15,24..31}
__device__ __forceinline__ bf16x16 load_a_frag(const bf16* A, int lda, int m0, int k0) {
  const int lane = threadIdx.x & 31;
  const int m  = m0 + (lane & 15);
  const int kb = k0 + ((lane & 16) ? 8 : 0);
  const bf16* row = A + m * lda + kb;
  return load_frag_2x16(row, row + 16);
}

// B fragment: 32x16 (KxN) from K-CONTIGUOUS (transposed) LDS tile Bt[n][k], ldk elements.
// lanes 0-15 -> N=lane, K{0..7,16..23} ; lanes 16-31 -> N=lane-16, K{8..15,24..31}
__device__ __forceinline__ bf16x16 load_bT_frag(const bf16* Bt, int ldk, int n0, int k0) {
  const int lane = threadIdx.x & 31;
  const int n  = n0 + (lane & 15);
  const int kb = k0 + ((lane & 16) ? 8 : 0);
  const bf16* p = Bt + n * ldk + kb;
  return load_frag_2x16(p, p + 16);
}

// ---- gfx1250 async global->LDS copy (16B per lane), tracked by ASYNCcnt ----
__device__ __forceinline__ void async_b128(void* lds, const void* g) {
  unsigned loff = (unsigned)(uintptr_t)lds;   // low 32 bits of LDS aperture addr = LDS offset
  asm volatile("global_load_async_to_lds_b128 %0, %1, off"
               :: "v"(loff), "v"(g) : "memory");
}
__device__ __forceinline__ void wait_async() {
  asm volatile("s_wait_asynccnt 0x0" ::: "memory");
}

// ---------------- 1) LayerNorm (with nan_to_num) -> bf16 ----------------
__global__ __launch_bounds__(256) void ln_kernel(const float* __restrict__ x,
                                                 const float* __restrict__ g,
                                                 const float* __restrict__ be,
                                                 bf16* __restrict__ xn) {
  const int row = blockIdx.x;
  const int tid = threadIdx.x;
  const f32x4* xr4 = reinterpret_cast<const f32x4*>(x + (size_t)row * D_);
  __shared__ float red[256];

  f32x4 local = xr4[tid];
  float s = 0.0f;
#pragma unroll
  for (int i = 0; i < 4; ++i) {
    float v = local[i];
    if (v != v) v = 0.0f;
    if (isinf(v)) v = (v > 0.f) ? 3.4028235e38f : -3.4028235e38f;
    local[i] = v;
    s += v;
  }
  red[tid] = s;
  __syncthreads();
  for (int off = 128; off > 0; off >>= 1) {
    if (tid < off) red[tid] += red[tid + off];
    __syncthreads();
  }
  const float mu = red[0] * (1.0f / D_);
  __syncthreads();

  float sq = 0.0f;
#pragma unroll
  for (int i = 0; i < 4; ++i) { float d = local[i] - mu; sq += d * d; }
  red[tid] = sq;
  __syncthreads();
  for (int off = 128; off > 0; off >>= 1) {
    if (tid < off) red[tid] += red[tid + off];
    __syncthreads();
  }
  const float inv = rsqrtf(red[0] * (1.0f / D_) + 1e-5f);

#pragma unroll
  for (int i = 0; i < 4; ++i) {
    const int c = tid * 4 + i;
    xn[(size_t)row * D_ + c] = f2bf((local[i] - mu) * inv * g[c] + be[c]);
  }
}

// ---------------- 2) f32 -> bf16 transposed weight convert ----------------
// dst[n*D + k] = (bf16) src[k*D + n]  -- makes B operands K-contiguous per N row
__global__ __launch_bounds__(256) void cvtT_kernel(const float* __restrict__ src,
                                                   bf16* __restrict__ dst) {
  __shared__ float tile[32][33];
  const int tx = threadIdx.x & 31, ty = threadIdx.x >> 5;  // 32 x 8
  const int k0 = blockIdx.x * 32, n0 = blockIdx.y * 32;
#pragma unroll
  for (int i = 0; i < 32; i += 8)
    tile[ty + i][tx] = src[(size_t)(k0 + ty + i) * D_ + n0 + tx];
  __syncthreads();
#pragma unroll
  for (int i = 0; i < 32; i += 8)
    dst[(size_t)(n0 + ty + i) * D_ + k0 + tx] = f2bf(tile[tx][ty + i]);
}

// ---------------- 3) QKV projection + eL contraction + clip ----------------
// grid: (BT/128, H). One head (64 N cols) per workgroup; 256 threads = 8 waves.
// Wt is transposed bf16 weight: Wt[n][k].
__global__ __launch_bounds__(256) void qkv_kernel(const bf16* __restrict__ xn,
                                                  const bf16* __restrict__ Wt,
                                                  const float* __restrict__ bias,
                                                  const float* __restrict__ eL,
                                                  bf16* __restrict__ out) {
  __shared__ __align__(16) bf16 sA[128 * 64];   // xn tile (row-major, lda=64)  16KB
  __shared__ __align__(16) bf16 sBt[64 * 64];   // Wt tile (K-contig per N row)  8KB
  __shared__ __align__(16) bf16 sELt[16 * 64];  // eL[h]^T : [r][d]              2KB
  __shared__ __align__(16) bf16 sP[8 * 16 * 64];// per-wave p staging           16KB

  const int tid   = threadIdx.x;
  const int wave  = tid >> 5;
  const int h     = blockIdx.y;
  const int mBase = blockIdx.x * 128;

  // eL[h] is (HD x R); store transposed: sELt[r*64 + d]
#pragma unroll
  for (int it = 0; it < 4; ++it) {
    const int i = tid + it * 256;          // i = d*16 + r (contiguous read)
    const int d = i >> 4, r = i & 15;
    sELt[r * 64 + d] = f2bf(eL[h * HD_ * R_ + i]);
  }

  floatx8 acc[4];
#pragma unroll
  for (int n = 0; n < 4; ++n) acc[n] = zero8();

  for (int kt = 0; kt < D_ / 64; ++kt) {
    const int k0 = kt * 64;
    __syncthreads();  // previous iteration's fragment reads complete WG-wide
    {
      const bf16* gA = xn + (size_t)mBase * D_ + k0;
#pragma unroll
      for (int it = 0; it < 4; ++it) {      // 128 rows * 128B = 1024 chunks
        const int c = tid + it * 256;
        const int row = c >> 3, sub = c & 7;
        async_b128(&sA[row * 64 + sub * 8], gA + (size_t)row * D_ + sub * 8);
      }
    }
    {
      const bf16* gB = Wt + (size_t)(h * HD_) * D_ + k0;
#pragma unroll
      for (int it = 0; it < 2; ++it) {      // 64 rows * 128B = 512 chunks
        const int c = tid + it * 256;
        const int row = c >> 3, sub = c & 7;
        async_b128(&sBt[row * 64 + sub * 8], gB + (size_t)row * D_ + sub * 8);
      }
    }
    wait_async();
    __syncthreads();
#pragma unroll
    for (int kk = 0; kk < 2; ++kk) {
      const bf16x16 a = load_a_frag(sA, 64, wave * 16, kk * 32);
#pragma unroll
      for (int n = 0; n < 4; ++n)
        acc[n] = wmma_bf16(a, load_bT_frag(sBt, 64, n * 16, kk * 32), acc[n]);
    }
  }

  // epilogue: p = acc + bias staged bf16 in LDS, then q = p @ eL[h], clip
  __syncthreads();
  {
    const int lane = tid & 31;
    const int mh = (lane & 16) ? 8 : 0;
    const int nl = lane & 15;
#pragma unroll
    for (int n = 0; n < 4; ++n) {
      const int col = n * 16 + nl;
      const float bsv = bias[h * HD_ + col];
#pragma unroll
      for (int i = 0; i < 8; ++i)
        sP[wave * 1024 + (mh + i) * 64 + col] = f2bf(acc[n][i] + bsv);
    }
  }
  __syncthreads();

  floatx8 qa = zero8();
  {
    const bf16* pw = sP + wave * 1024;  // own 16x64 tile
#pragma unroll
    for (int kk = 0; kk < 2; ++kk) {
      const bf16x16 a = load_a_frag(pw, 64, 0, kk * 32);
      const bf16x16 b = load_bT_frag(sELt, 64, 0, kk * 32);
      qa = wmma_bf16(a, b, qa);
    }
  }
  {
    const int lane = tid & 31;
    const int mh = (lane & 16) ? 8 : 0;
    const int r = lane & 15;
#pragma unroll
    for (int i = 0; i < 8; ++i) {
      float v = qa[i];
      v = fminf(10.0f, fmaxf(-10.0f, v));
      const int row = mBase + wave * 16 + mh + i;
      out[(size_t)row * HR_ + h * R_ + r] = f2bf(v);
    }
  }
}

// ---------------- 4) scores over T, softmax, Weff = probs^T @ eR ----------------
__global__ __launch_bounds__(256) void scores_kernel(const bf16* __restrict__ q,
                                                     const bf16* __restrict__ k,
                                                     const float* __restrict__ eR,
                                                     float* __restrict__ weff) {
  const int bh = blockIdx.x;
  const int b = bh / H_, h = bh % H_;
  __shared__ __align__(16) bf16 sq[128 * 16];
  __shared__ __align__(16) bf16 sk[128 * 16];
  __shared__ float sprob[16 * 16];

  const int tid = threadIdx.x;
  const int r = tid >> 4, s = tid & 15;

  float acc = 0.0f;
  for (int t0 = 0; t0 < T_; t0 += 128) {
    __syncthreads();
    {
      const int rr = tid & 127;
      const bf16* src = ((tid < 128) ? q : k) +
                        ((size_t)(b * T_ + t0 + rr) * HR_ + h * R_);
      bf16* dst = ((tid < 128) ? sq : sk) + rr * 16;
      const u32x4* s4 = reinterpret_cast<const u32x4*>(src);
      u32x4* d4 = reinterpret_cast<u32x4*>(dst);
      d4[0] = s4[0];
      d4[1] = s4[1];
    }
    __syncthreads();
#pragma unroll 8
    for (int tt = 0; tt < 128; ++tt)
      acc += bf2f(sq[tt * 16 + r]) * bf2f(sk[tt * 16 + s]);
  }
  acc *= 0.25f;  // 1/sqrt(R)

  __syncthreads();
  sprob[r * 16 + s] = acc;
  __syncthreads();
  float mx = -3.4028235e38f;
#pragma unroll
  for (int j = 0; j < 16; ++j) mx = fmaxf(mx, sprob[r * 16 + j]);
  float sum = 0.0f;
#pragma unroll
  for (int j = 0; j < 16; ++j) sum += __expf(sprob[r * 16 + j] - mx);
  const float p = __expf(acc - mx) / sum;
  __syncthreads();
  sprob[r * 16 + s] = p;
  __syncthreads();

  // Weff[s][d] = sum_r probs[r][s] * eR[h][r][d]
  for (int o = tid; o < R_ * HD_; o += 256) {
    const int ss = o / HD_, d = o % HD_;
    float a2 = 0.0f;
#pragma unroll
    for (int rr = 0; rr < 16; ++rr)
      a2 += sprob[rr * 16 + ss] * eR[(size_t)(h * R_ + rr) * HD_ + d];
    weff[(size_t)bh * R_ * HD_ + o] = a2;
  }
}

// ---------------- 5) WfoldT[b][j][h*16+s] = sum_d Weff[b,h][s][d]*Wp[h*64+d][j] ----------------
// stored transposed (j-major rows of 256 K values) for b128 tile staging in out GEMM
__global__ __launch_bounds__(256) void wfold_kernel(const float* __restrict__ weff,
                                                    const float* __restrict__ Wp,
                                                    bf16* __restrict__ wfoldT) {
  const int bh = blockIdx.x;
  const int b = bh / H_, h = bh % H_;
  __shared__ float sW[R_ * HD_];
  const int tid = threadIdx.x;
  for (int i = tid; i < R_ * HD_; i += 256) sW[i] = weff[(size_t)bh * R_ * HD_ + i];
  __syncthreads();

  for (int o = tid; o < R_ * D_; o += 256) {
    const int s = o / D_, j = o % D_;
    float acc = 0.0f;
#pragma unroll 16
    for (int d = 0; d < HD_; ++d)
      acc += sW[s * HD_ + d] * Wp[(size_t)(h * HD_ + d) * D_ + j];
    wfoldT[((size_t)b * D_ + j) * HR_ + h * R_ + s] = f2bf(acc);
  }
}

// ---------------- 6) out = v(BT x 256) @ Wfold[b](256 x 1024) + bp ----------------
// grid: (BT/128, D/64); WfoldT[b][j][k] is K-contiguous.
__global__ __launch_bounds__(256) void out_kernel(const bf16* __restrict__ v,
                                                  const bf16* __restrict__ wfoldT,
                                                  const float* __restrict__ bp,
                                                  float* __restrict__ out) {
  __shared__ __align__(16) bf16 sA[128 * 64];  // v tile, lda=64
  __shared__ __align__(16) bf16 sBt[64 * 64];  // WfoldT tile, K-contig per N row

  const int tid = threadIdx.x;
  const int wave = tid >> 5;
  const int mBase = blockIdx.x * 128;
  const int n0 = blockIdx.y * 64;
  const int b = mBase / T_;

  floatx8 acc[4];
#pragma unroll
  for (int n = 0; n < 4; ++n) acc[n] = zero8();

  for (int kt = 0; kt < HR_ / 64; ++kt) {
    const int k0 = kt * 64;
    __syncthreads();
    {
      const bf16* gA = v + (size_t)mBase * HR_ + k0;
#pragma unroll
      for (int it = 0; it < 4; ++it) {
        const int c = tid + it * 256;
        const int row = c >> 3, sub = c & 7;
        async_b128(&sA[row * 64 + sub * 8], gA + (size_t)row * HR_ + sub * 8);
      }
    }
    {
      const bf16* gB = wfoldT + ((size_t)b * D_ + n0) * HR_ + k0;
#pragma unroll
      for (int it = 0; it < 2; ++it) {
        const int c = tid + it * 256;
        const int row = c >> 3, sub = c & 7;
        async_b128(&sBt[row * 64 + sub * 8], gB + (size_t)row * HR_ + sub * 8);
      }
    }
    wait_async();
    __syncthreads();
#pragma unroll
    for (int kk = 0; kk < 2; ++kk) {
      const bf16x16 a = load_a_frag(sA, 64, wave * 16, kk * 32);
#pragma unroll
      for (int n = 0; n < 4; ++n)
        acc[n] = wmma_bf16(a, load_bT_frag(sBt, 64, n * 16, kk * 32), acc[n]);
    }
  }

  const int lane = tid & 31;
  const int mh = (lane & 16) ? 8 : 0;
  const int nl = lane & 15;
#pragma unroll
  for (int n = 0; n < 4; ++n) {
    const int col = n0 + n * 16 + nl;
    const float bias = bp[col];
#pragma unroll
    for (int i = 0; i < 8; ++i) {
      const int row = mBase + wave * 16 + mh + i;
      float val = acc[n][i] + bias;
      if (val != val) val = 0.0f;  // nan_to_num
      out[(size_t)row * D_ + col] = val;
    }
  }
}

// ---------------- host launcher ----------------
extern "C" void kernel_launch(void* const* d_in, const int* in_sizes, int n_in,
                              void* d_out, int out_size, void* d_ws, size_t ws_size,
                              hipStream_t stream) {
  const float* x    = (const float*)d_in[0];
  const float* ln_g = (const float*)d_in[1];
  const float* ln_b = (const float*)d_in[2];
  const float* Wq   = (const float*)d_in[3];
  const float* bq   = (const float*)d_in[4];
  const float* Wk   = (const float*)d_in[5];
  const float* bk   = (const float*)d_in[6];
  const float* Wv   = (const float*)d_in[7];
  const float* bv   = (const float*)d_in[8];
  const float* eL   = (const float*)d_in[9];
  const float* eR   = (const float*)d_in[10];
  const float* Wp   = (const float*)d_in[11];
  const float* bp   = (const float*)d_in[12];
  float* out = (float*)d_out;

  char* ws = (char*)d_ws;
  size_t off = 0;
  auto alloc = [&](size_t bytes) {
    char* p = ws + off;
    off = (off + bytes + 255) & ~(size_t)255;
    return p;
  };
  bf16* xn    = (bf16*)alloc((size_t)BT_ * D_ * sizeof(bf16));   // 32 MB
  bf16* WqT   = (bf16*)alloc((size_t)D_ * D_ * sizeof(bf16));    // 2 MB each
  bf16* WkT   = (bf16*)alloc((size_t)D_ * D_ * sizeof(bf16));
  bf16* WvT   = (bf16*)alloc((size_t)D_ * D_ * sizeof(bf16));
  bf16* qb    = (bf16*)alloc((size_t)BT_ * HR_ * sizeof(bf16));  // 8 MB each
  bf16* kb    = (bf16*)alloc((size_t)BT_ * HR_ * sizeof(bf16));
  bf16* vb    = (bf16*)alloc((size_t)BT_ * HR_ * sizeof(bf16));
  float* weff = (float*)alloc((size_t)B_ * H_ * R_ * HD_ * sizeof(float));
  bf16* wfoldT = (bf16*)alloc((size_t)B_ * D_ * HR_ * sizeof(bf16));
  (void)ws_size; (void)in_sizes; (void)n_in; (void)out_size;

  // 1) layernorm
  ln_kernel<<<dim3(BT_), dim3(256), 0, stream>>>(x, ln_g, ln_b, xn);

  // 2) transposed weight conversion (B operands become K-contiguous)
  dim3 gt(D_ / 32, D_ / 32);
  cvtT_kernel<<<gt, dim3(256), 0, stream>>>(Wq, WqT);
  cvtT_kernel<<<gt, dim3(256), 0, stream>>>(Wk, WkT);
  cvtT_kernel<<<gt, dim3(256), 0, stream>>>(Wv, WvT);

  // 3) q/k/v projections (+eL, clip)
  dim3 gp(BT_ / 128, H_);
  qkv_kernel<<<gp, dim3(256), 0, stream>>>(xn, WqT, bq, eL, qb);
  qkv_kernel<<<gp, dim3(256), 0, stream>>>(xn, WkT, bk, eL, kb);
  qkv_kernel<<<gp, dim3(256), 0, stream>>>(xn, WvT, bv, eL, vb);

  // 4) scores + softmax + Weff
  scores_kernel<<<dim3(B_ * H_), dim3(256), 0, stream>>>(qb, kb, eR, weff);

  // 5) fold Weff into Wp (transposed output)
  wfold_kernel<<<dim3(B_ * H_), dim3(256), 0, stream>>>(weff, Wp, wfoldT);

  // 6) final GEMM (K folded to 256)
  out_kernel<<<dim3(BT_ / 128, D_ / 64), dim3(256), 0, stream>>>(vb, wfoldT, bp, out);
}
